// GCN_14173392077042
// MI455X (gfx1250) — compile-verified, matrix operand order checked by reading, not run on hardware
//
#include <hip/hip_runtime.h>
#include <hip/hip_bf16.h>

#define NNODES 100000
#define NEDGES 1600000
#define D 128
#define BN_EPS 1e-5f

typedef __attribute__((ext_vector_type(16))) _Float16 v16h;
typedef __attribute__((ext_vector_type(8)))  float    v8f;

// ---------------------------------------------------------------- utilities
__global__ void gcn_zero_f32(float* __restrict__ p, long n) {
    long i = (long)blockIdx.x * blockDim.x + threadIdx.x;
    long stride = (long)gridDim.x * blockDim.x;
    for (; i < n; i += stride) p[i] = 0.0f;
}

__global__ void gcn_degree(const int* __restrict__ src, const int* __restrict__ dst,
                           float* __restrict__ deg_out, float* __restrict__ deg_in, int e_cnt) {
    int e = blockIdx.x * blockDim.x + threadIdx.x;
    if (e >= e_cnt) return;
    atomicAdd(&deg_out[src[e]], 1.0f);
    atomicAdd(&deg_in[dst[e]], 1.0f);
}

__global__ void gcn_rsqrt_deg(float* __restrict__ iso, float* __restrict__ isi, int n) {
    int i = blockIdx.x * blockDim.x + threadIdx.x;
    if (i >= n) return;
    iso[i] = rsqrtf(fmaxf(iso[i], 1.0f));
    isi[i] = rsqrtf(fmaxf(isi[i], 1.0f));
}

// h_scaled[row][:] = h[row][:] * iso[row]   (float4 vectorized, 32 float4 per row)
__global__ void gcn_scale_rows(const float* __restrict__ h, const float* __restrict__ iso,
                               float* __restrict__ out, int n) {
    int t = blockIdx.x * blockDim.x + threadIdx.x;     // one float4 per thread
    if (t >= n * 32) return;
    int row = t >> 5;
    float s = iso[row];
    float4 v = ((const float4*)h)[t];
    v.x *= s; v.y *= s; v.z *= s; v.w *= s;
    ((float4*)out)[t] = v;
}

// one wave per edge: lane handles 4 channels; bandwidth-bound scatter-add
__global__ __launch_bounds__(256) void gcn_scatter(const float* __restrict__ hs,
                                                   const int* __restrict__ src,
                                                   const int* __restrict__ dst,
                                                   float* __restrict__ agg, int e_cnt) {
    int wave = threadIdx.x >> 5;
    int lane = threadIdx.x & 31;
    int e = blockIdx.x * 8 + wave;
    if (e >= e_cnt) return;
    int s = src[e], d = dst[e];
    float4 v = ((const float4*)(hs + (size_t)s * D))[lane];
    float* dp = agg + (size_t)d * D + lane * 4;
    __builtin_prefetch(dp, 0, 1);                      // global_prefetch_b8 on the dest line
    atomicAdd(dp + 0, v.x);
    atomicAdd(dp + 1, v.y);
    atomicAdd(dp + 2, v.z);
    atomicAdd(dp + 3, v.w);
}

// Pre-pack W (f32, row-major K x N = 128x128) into WMMA B-fragment layout:
//   wp[((tile_j*4 + kb)*32 + lane)*16 + i]  -> one contiguous v16h per (tile,kb,lane)
// Per the ISA 16-bit 32x16 B layout: lane l (n = l&15, hl = l>>4) supplies
//   i in [0,8):  K = kb*32 + hl*8 + i
//   i in [8,16): K = kb*32 + 16 + hl*8 + (i-8)
__global__ void gcn_pack_w(const float* __restrict__ w, _Float16* __restrict__ wp) {
    int t = blockIdx.x * blockDim.x + threadIdx.x;     // 8*4*32 = 1024 fragments
    if (t >= 8 * 4 * 32) return;
    int lane = t & 31;
    int kbi  = (t >> 5) & 3;
    int tj   = t >> 7;
    int hl = lane >> 4;
    int col = tj * 16 + (lane & 15);
    _Float16* o = wp + (size_t)t * 16;
#pragma unroll
    for (int i = 0; i < 8; ++i) {
        int k0 = kbi * 32 + hl * 8 + i;
        int k1 = k0 + 16;
        o[i]     = (_Float16)w[k0 * D + col];
        o[8 + i] = (_Float16)w[k1 * D + col];
    }
}

// ---------------------------------------------------------------- WMMA GEMM
// T[i][j] = sum_k (agg[i][k]*isi[i]) * W[k][j] + bias[j] + hlast[i][j]
// block = 256 threads = 8 waves; block covers 16 rows, wave w covers cols [16w,16w+16)
// K = 128 -> 4 chained v_wmma_f32_16x16x32_f16; B comes pre-packed per-lane.
__global__ __launch_bounds__(256) void gcn_gemm_wmma(
    const float* __restrict__ agg, const float* __restrict__ isi,
    const _Float16* __restrict__ wp, const float* __restrict__ bias,
    const float* __restrict__ hlast, float* __restrict__ T,
    float* __restrict__ colsum, float* __restrict__ colsumsq, int with_stats) {

    const int wave = threadIdx.x >> 5;
    const int lane = threadIdx.x & 31;
    const int hl   = lane >> 4;          // which half of the wave
    const int m    = lane & 15;          // A row this lane supplies / D col this lane holds
    const int i0   = blockIdx.x * 16;
    const int j0   = wave * 16;

    const int rowA = i0 + m;
    const float isir = isi[rowA];
    const float* arow = agg + (size_t)rowA * D;
    const v16h* bfrag = (const v16h*)wp + (size_t)wave * 4 * 32 + lane;

    v8f c = {};
#pragma unroll
    for (int kbi = 0; kbi < 4; ++kbi) {
        const int kb = kbi * 32;
        v16h a;
#pragma unroll
        for (int i = 0; i < 8; ++i) {
            int k0 = kb + hl * 8 + i;        // K in [kb, kb+16)
            a[i]     = (_Float16)(arow[k0] * isir);
            a[8 + i] = (_Float16)(arow[k0 + 16] * isir);
        }
        v16h b = bfrag[kbi * 32];            // contiguous 32B per lane, coalesced
        c = __builtin_amdgcn_wmma_f32_16x16x32_f16(
                /*neg_a=*/false, a, /*neg_b=*/false, b,
                /*c_mod=*/(short)0, c, /*reuse_a=*/false, /*reuse_b=*/false);
    }

    // epilogue: bias + residual, store, and per-column BN partial sums
    const float bj = bias[j0 + m];
    float s = 0.0f, ss = 0.0f;
#pragma unroll
    for (int r = 0; r < 8; ++r) {
        int row = i0 + hl * 8 + r;           // C/D layout: lanes>=16 hold rows M=8..15
        float v = c[r] + bj + hlast[(size_t)row * D + j0 + m];
        T[(size_t)row * D + j0 + m] = v;
        s += v; ss += v * v;
    }
    if (with_stats) {
        s  += __shfl_down(s, 16, 32);        // combine the two lanes sharing a column
        ss += __shfl_down(ss, 16, 32);
        if (lane < 16) {
            atomicAdd(&colsum[j0 + lane], s);
            atomicAdd(&colsumsq[j0 + lane], ss);
        }
    }
}

__global__ void gcn_bn_finalize(const float* __restrict__ colsum, const float* __restrict__ colsumsq,
                                const float* __restrict__ g, const float* __restrict__ beta,
                                float* __restrict__ scl, float* __restrict__ shf, float inv_n) {
    int c = threadIdx.x;                     // 128 threads
    float mu  = colsum[c] * inv_n;
    float var = colsumsq[c] * inv_n - mu * mu;
    float rs  = rsqrtf(var + BN_EPS) * g[c];
    scl[c] = rs;
    shf[c] = beta[c] - mu * rs;
}

__global__ void gcn_bn_relu(float* __restrict__ t, const float* __restrict__ scl,
                            const float* __restrict__ shf, long n) {
    long i = (long)blockIdx.x * blockDim.x + threadIdx.x;
    if (i >= n) return;
    int c = (int)(i & (D - 1));
    t[i] = fmaxf(t[i] * scl[c] + shf[c], 0.0f);
}

// ---------------------------------------------------------------- driver
extern "C" void kernel_launch(void* const* d_in, const int* in_sizes, int n_in,
                              void* d_out, int out_size, void* d_ws, size_t ws_size,
                              hipStream_t stream) {
    const float* feat = (const float*)d_in[0];
    const int*   src  = (const int*)d_in[1];
    const int*   dst  = (const int*)d_in[2];
    const float* W[3]    = { (const float*)d_in[3], (const float*)d_in[5], (const float*)d_in[7] };
    const float* bvec[3] = { (const float*)d_in[4], (const float*)d_in[6], (const float*)d_in[8] };
    const float* g[2]    = { (const float*)d_in[9],  (const float*)d_in[11] };
    const float* beta[2] = { (const float*)d_in[10], (const float*)d_in[12] };

    const long NF = (long)NNODES * D;

    // workspace carving
    float* iso      = (float*)d_ws;                 // N  (degree -> rsqrt, in place)
    float* isi      = iso + NNODES;                 // N
    float* colsum   = isi + NNODES;                 // 128
    float* colsumsq = colsum + D;                   // 128
    float* scl      = colsumsq + D;                 // 128
    float* shf      = scl + D;                      // 128
    _Float16* wp    = (_Float16*)(shf + D);         // 128*128 halves (32 KB), 32B-aligned
    float* B0       = (float*)(wp + D * D);         // N*128
    float* B1       = B0 + NF;                      // N*128
    float* B2       = B1 + NF;                      // N*128

    const int TPB = 256;
    const dim3 blk(TPB);
    const int grid_e    = (NEDGES + TPB - 1) / TPB;
    const int grid_n    = (NNODES + TPB - 1) / TPB;
    const int grid_v4   = (NNODES * 32 + TPB - 1) / TPB;   // float4 row-scale
    const int grid_sc   = (NEDGES + 7) / 8;                // wave-per-edge scatter
    const int grid_mm   = NNODES / 16;                     // 6250 WMMA row tiles
    const int grid_elem = (int)((NF + TPB - 1) / TPB);

    // degrees -> iso/isi
    gcn_zero_f32<<<2048, blk, 0, stream>>>(iso, 2L * NNODES);
    gcn_degree<<<grid_e, blk, 0, stream>>>(src, dst, iso, isi, NEDGES);
    gcn_rsqrt_deg<<<grid_n, blk, 0, stream>>>(iso, isi, NNODES);

    // ---------------- layer 0 : hlast = feat, scaled->B0, agg->B1, T->B0
    gcn_scale_rows<<<grid_v4, blk, 0, stream>>>(feat, iso, B0, NNODES);
    gcn_zero_f32<<<4096, blk, 0, stream>>>(B1, NF);
    gcn_scatter<<<grid_sc, blk, 0, stream>>>(B0, src, dst, B1, NEDGES);
    gcn_pack_w<<<4, blk, 0, stream>>>(W[0], wp);
    gcn_zero_f32<<<1, blk, 0, stream>>>(colsum, 2 * D);
    gcn_gemm_wmma<<<grid_mm, blk, 0, stream>>>(B1, isi, wp, bvec[0], feat, B0,
                                               colsum, colsumsq, 1);
    gcn_bn_finalize<<<1, D, 0, stream>>>(colsum, colsumsq, g[0], beta[0], scl, shf,
                                         1.0f / (float)NNODES);
    gcn_bn_relu<<<grid_elem, blk, 0, stream>>>(B0, scl, shf, NF);

    // ---------------- layer 1 : hlast = B0, scaled->B1, agg->B2, T->B1
    gcn_scale_rows<<<grid_v4, blk, 0, stream>>>(B0, iso, B1, NNODES);
    gcn_zero_f32<<<4096, blk, 0, stream>>>(B2, NF);
    gcn_scatter<<<grid_sc, blk, 0, stream>>>(B1, src, dst, B2, NEDGES);
    gcn_pack_w<<<4, blk, 0, stream>>>(W[1], wp);
    gcn_zero_f32<<<1, blk, 0, stream>>>(colsum, 2 * D);
    gcn_gemm_wmma<<<grid_mm, blk, 0, stream>>>(B2, isi, wp, bvec[1], B0, B1,
                                               colsum, colsumsq, 1);
    gcn_bn_finalize<<<1, D, 0, stream>>>(colsum, colsumsq, g[1], beta[1], scl, shf,
                                         1.0f / (float)NNODES);
    gcn_bn_relu<<<grid_elem, blk, 0, stream>>>(B1, scl, shf, NF);

    // ---------------- layer 2 : hlast = B1, scaled->B2, agg->B0, out -> d_out
    gcn_scale_rows<<<grid_v4, blk, 0, stream>>>(B1, iso, B2, NNODES);
    gcn_zero_f32<<<4096, blk, 0, stream>>>(B0, NF);
    gcn_scatter<<<grid_sc, blk, 0, stream>>>(B2, src, dst, B0, NEDGES);
    gcn_pack_w<<<4, blk, 0, stream>>>(W[2], wp);
    gcn_gemm_wmma<<<grid_mm, blk, 0, stream>>>(B0, isi, wp, bvec[2], B1, (float*)d_out,
                                               colsum, colsumsq, 0);
}
